// BidirLayer_71167608095136
// MI455X (gfx1250) — matrix operand order repaired and checked by reading, not compiled, not run on hardware
//
#include <hip/hip_runtime.h>
#include <hip/hip_bf16.h>
#include <stdint.h>

// Problem constants (match reference)
#define TT 512
#define BB 64
#define II 1024
#define HH 1024
#define G4 4096   // 4*H
#define OH 2048   // 2*H

typedef __attribute__((ext_vector_type(16))) __bf16 v16bf;
typedef __attribute__((ext_vector_type(8)))  __bf16 v8bf;
typedef __attribute__((ext_vector_type(8)))  float  v8f;

__device__ __forceinline__ unsigned short f32_to_bf16(float f) {
  union { float f; unsigned int u; } v; v.f = f;
  unsigned int lsb = (v.u >> 16) & 1u;
  return (unsigned short)((v.u + 0x7FFFu + lsb) >> 16);
}

__device__ __forceinline__ float sigmoidf(float x) {
  return 1.0f / (1.0f + __expf(-x));
}

// ---------------------------------------------------------------------------
// Wt[k][4H] (bf16) = W[4H][K] (f32) transposed + converted.  Output-coalesced.
// ---------------------------------------------------------------------------
__global__ void wcvt_kernel(const float* __restrict__ W,
                            unsigned short* __restrict__ Wt,
                            int rows /*4096*/, int cols /*K*/) {
  int idx = blockIdx.x * blockDim.x + threadIdx.x;
  int total = rows * cols;
  if (idx < total) {
    int r = idx % rows;   // original row (gate-column in Wt) -> fast dim
    int k = idx / rows;
    Wt[(size_t)k * rows + r] = f32_to_bf16(W[(size_t)r * cols + k]);
  }
}

// ---------------------------------------------------------------------------
// Bulk fp32 -> bf16 convert (x tensor)
// ---------------------------------------------------------------------------
__global__ void xcvt_kernel(const float* __restrict__ x,
                            unsigned short* __restrict__ xb, size_t n) {
  size_t i = (size_t)blockIdx.x * blockDim.x + threadIdx.x;
  if (i < n) xb[i] = f32_to_bf16(x[i]);
}

// ---------------------------------------------------------------------------
// Init recurrent state: h (bf16, parity 0 for both dirs), c (fp32)
// ---------------------------------------------------------------------------
__global__ void init_state_kernel(const float* __restrict__ h0f,
                                  const float* __restrict__ c0f,
                                  const float* __restrict__ h0r,
                                  const float* __restrict__ c0r,
                                  unsigned short* __restrict__ hbuf, // [2][2][B][H]
                                  float* __restrict__ cbuf) {        // [2][B][H]
  int i = blockIdx.x * blockDim.x + threadIdx.x;
  const int n = BB * HH;
  if (i < n) {
    hbuf[i]                 = f32_to_bf16(h0f[i]);       // dir0, parity0
    hbuf[(size_t)2 * n + i] = f32_to_bf16(h0r[i]);       // dir1, parity0
    cbuf[i]                 = c0f[i];
    cbuf[(size_t)n + i]     = c0r[i];
  }
}

// ---------------------------------------------------------------------------
// One recurrence step, both directions. Fused:
//   proj = sigmoid(x_t @ W_ih^T + h @ W_hh^T + b)   via v_wmma_f32_16x16x32_bf16
//   c' = c*f + z - i ; h' = sigmoid(c') - o
//
// grid = (H/32, 2 dirs), block = 256 (8 waves).
// Wave w: batch-tile mt = w&3, j-subtile jt = w>>2.
// Each wave keeps FOUR 16x16 f32 accumulators (one per gate) so the A
// fragment (x_t / h rows) is loaded ONCE per k-chunk and feeds 4 WMMAs.
// ---------------------------------------------------------------------------
__global__ __launch_bounds__(256)
void step_kernel(const unsigned short* __restrict__ xbf,    // [T][B][I] bf16
                 const unsigned short* __restrict__ WtihF,  // [I][4H]
                 const unsigned short* __restrict__ WthhF,  // [H][4H]
                 const float* __restrict__ bF,
                 const unsigned short* __restrict__ WtihR,
                 const unsigned short* __restrict__ WthhR,
                 const float* __restrict__ bR,
                 unsigned short* __restrict__ hbuf,         // [2][2][B][H] bf16
                 float* __restrict__ cbuf,                  // [2][B][H]
                 float* __restrict__ out,                   // [T][B][2H]
                 int s) {
  const int dir = blockIdx.y;
  const int jb  = blockIdx.x * 32;     // 32 hidden units per block
  const int t   = dir ? (TT - 1 - s) : s;
  const int par = s & 1;
  const int SN  = BB * HH;

  const unsigned short* Wtih = dir ? WtihR : WtihF;
  const unsigned short* Wthh = dir ? WthhR : WthhF;
  const float*          bias = dir ? bR : bF;
  const unsigned short* xrow = xbf + (size_t)t * BB * II;
  const unsigned short* hrow = hbuf + ((size_t)dir * 2 + par) * SN;
  unsigned short*      hnext = hbuf + ((size_t)dir * 2 + (par ^ 1)) * SN;
  float*                crow = cbuf + (size_t)dir * SN;

  __shared__ float gates[4][BB][32];   // 32 KB gate exchange

  const int wave = threadIdx.x >> 5;   // 0..7
  const int lane = threadIdx.x & 31;
  const int mt   = wave & 3;           // batch tile 0..3
  const int jt   = wave >> 2;          // j-subtile 0..1
  const int j0   = jb + jt * 16;

  // A-fragment addressing (16-bit A 16x32 wave32 layout):
  //   lane<16: row m=lane, K = {0..7, 16..23}; lane>=16: row m=lane-16, K += 8
  const int m    = lane & 15;
  const int row  = mt * 16 + m;
  const int koff = (lane >> 4) ? 8 : 0;

  v8f acc[4] = {v8f{}, v8f{}, v8f{}, v8f{}};
  #pragma unroll 1
  for (int ph = 0; ph < 2; ++ph) {
    const unsigned short* A  = ph ? hrow : xrow;   // [B][1024] bf16 row-major
    const unsigned short* Bm = ph ? Wthh : Wtih;   // [1024][4096] bf16
    #pragma unroll 2
    for (int k0 = 0; k0 < 1024; k0 += 32) {
      union { v16bf v; v8bf h[2]; } a;
      const unsigned short* ap = A + (size_t)row * 1024 + k0 + koff;
      a.h[0] = *(const v8bf*)(ap);          // K = k0+koff .. +7
      a.h[1] = *(const v8bf*)(ap + 16);     // K = k0+koff+16 .. +23
      // B-fragment: lane = N column (j0+m), K = same table order as A
      const unsigned short* bp = Bm + (size_t)(k0 + koff) * G4 + j0 + m;
      #pragma unroll
      for (int g = 0; g < 4; ++g) {
        union { v16bf v; unsigned short u[16]; } b;
        #pragma unroll
        for (int i = 0; i < 16; ++i)
          b.u[i] = bp[g * HH + (size_t)((i & 7) + ((i >> 3) << 4)) * G4];
        acc[g] = __builtin_amdgcn_wmma_f32_16x16x32_bf16(
                     false, a.v, false, b.v, (short)0, acc[g], false, false);
      }
    }
  }

  // C/D layout: lane n = lane&15; rows m = (lane>=16 ? 8 : 0) + r
  const int n  = lane & 15;
  const int mbase = mt * 16 + ((lane >> 4) ? 8 : 0);
  #pragma unroll
  for (int g = 0; g < 4; ++g) {
    const float bv = bias[g * HH + j0 + n];
    #pragma unroll
    for (int r = 0; r < 8; ++r) {
      gates[g][mbase + r][jt * 16 + n] = sigmoidf(acc[g][r] + bv);
    }
  }
  __syncthreads();

  // Cell update: B*32 = 2048 elements across 256 threads
  for (int e = threadIdx.x; e < BB * 32; e += 256) {
    const int bb = e >> 5;
    const int jj = e & 31;
    const float ig = gates[0][bb][jj];
    const float og = gates[1][bb][jj];
    const float zg = gates[2][bb][jj];
    const float fg = gates[3][bb][jj];
    const size_t cidx = (size_t)bb * HH + jb + jj;
    const float c  = crow[cidx];
    const float cn = c * fg + zg - ig;
    const float hn = sigmoidf(cn) - og;
    crow[cidx]  = cn;
    hnext[cidx] = f32_to_bf16(hn);
    out[((size_t)t * BB + bb) * OH + (size_t)dir * HH + jb + jj] = hn;
  }
}

// ---------------------------------------------------------------------------
// Final states: hf = out[T-1,:,0:H], hr = out[0,:,H:2H], cf/cr from cbuf
// ---------------------------------------------------------------------------
__global__ void finalize_kernel(const float* __restrict__ out,
                                const float* __restrict__ cbuf,
                                float* __restrict__ hf, float* __restrict__ cf,
                                float* __restrict__ hr, float* __restrict__ cr) {
  int i = blockIdx.x * blockDim.x + threadIdx.x;
  const int nBH = BB * HH;
  if (i < nBH) {
    const int b = i / HH, j = i % HH;
    hf[i] = out[((size_t)(TT - 1) * BB + b) * OH + j];
    hr[i] = out[(size_t)b * OH + HH + j];
    cf[i] = cbuf[i];
    cr[i] = cbuf[(size_t)nBH + i];
  }
}

extern "C" void kernel_launch(void* const* d_in, const int* in_sizes, int n_in,
                              void* d_out, int out_size, void* d_ws, size_t ws_size,
                              hipStream_t stream) {
  const float* x     = (const float*)d_in[0];
  const float* h0f   = (const float*)d_in[1];
  const float* c0f   = (const float*)d_in[2];
  const float* h0r   = (const float*)d_in[3];
  const float* c0r   = (const float*)d_in[4];
  const float* WihF  = (const float*)d_in[5];
  const float* WhhF  = (const float*)d_in[6];
  const float* bF    = (const float*)d_in[7];
  const float* WihR  = (const float*)d_in[8];
  const float* WhhR  = (const float*)d_in[9];
  const float* bR    = (const float*)d_in[10];

  // Workspace layout (~97 MB):
  //  [0,8MB)    Wt_ih_f bf16 [I][4H]
  //  [8,16MB)   Wt_hh_f
  //  [16,24MB)  Wt_ih_r
  //  [24,32MB)  Wt_hh_r
  //  [32,96MB)  x bf16 [T][B][I]
  //  [96MB,+512KB) h double-buffer bf16 [2 dir][2 parity][B][H]
  //  then       c fp32 [2 dir][B][H]
  char* ws = (char*)d_ws;
  const size_t MB = 1024ull * 1024ull;
  unsigned short* WtihF = (unsigned short*)(ws + 0 * MB);
  unsigned short* WthhF = (unsigned short*)(ws + 8 * MB);
  unsigned short* WtihR = (unsigned short*)(ws + 16 * MB);
  unsigned short* WthhR = (unsigned short*)(ws + 24 * MB);
  unsigned short* xbf   = (unsigned short*)(ws + 32 * MB);
  unsigned short* hbuf  = (unsigned short*)(ws + 96 * MB);
  float*          cbuf  = (float*)(ws + 96 * MB + 512 * 1024);

  // Output layout: out [T][B][2H], then hf, cf, hr, cr (each B*H)
  float* out = (float*)d_out;
  float* hf  = out + (size_t)TT * BB * OH;
  float* cf  = hf + (size_t)BB * HH;
  float* hr  = cf + (size_t)BB * HH;
  float* cr  = hr + (size_t)BB * HH;

  // 1) Weight convert + transpose to [K][4H] bf16 (once per call, L2-resident)
  {
    const int total = G4 * II;
    const int blocks = (total + 255) / 256;
    wcvt_kernel<<<blocks, 256, 0, stream>>>(WihF, WtihF, G4, II);
    wcvt_kernel<<<blocks, 256, 0, stream>>>(WhhF, WthhF, G4, HH);
    wcvt_kernel<<<blocks, 256, 0, stream>>>(WihR, WtihR, G4, II);
    wcvt_kernel<<<blocks, 256, 0, stream>>>(WhhR, WthhR, G4, HH);
  }
  // 2) x -> bf16
  {
    const size_t n = (size_t)TT * BB * II;
    xcvt_kernel<<<(unsigned)((n + 255) / 256), 256, 0, stream>>>(x, xbf, n);
  }
  // 3) initial state
  init_state_kernel<<<(BB * HH + 255) / 256, 256, 0, stream>>>(
      h0f, c0f, h0r, c0r, hbuf, cbuf);

  // 4) sequential scan: fused dual-direction WMMA step per timestep
  dim3 grid(HH / 32, 2);
  for (int s = 0; s < TT; ++s) {
    step_kernel<<<grid, 256, 0, stream>>>(xbf, WtihF, WthhF, bF,
                                          WtihR, WthhR, bR,
                                          hbuf, cbuf, out, s);
  }
  // 5) final states
  finalize_kernel<<<(BB * HH + 255) / 256, 256, 0, stream>>>(
      out, cbuf, hf, cf, hr, cr);
}
